// SpatialMambaBlock_85856396247379
// MI455X (gfx1250) — matrix operand order repaired and checked by reading, not compiled, not run on hardware
//
#include <hip/hip_runtime.h>
#include <hip/hip_bf16.h>

// ---------- types ----------
typedef __attribute__((ext_vector_type(16))) __bf16 v16bf;
typedef __attribute__((ext_vector_type(8)))  __bf16 v8bf;
typedef __attribute__((ext_vector_type(8)))  float  v8f;

// ---------- problem dims ----------
#define BATCH 8
#define CCH   192
#define LL    1024
#define BL    (BATCH * LL)      // 8192
#define DIN   384
#define NXZ   (2 * DIN)         // 768
#define NST   16
#define DTR   12
#define NDBL  48                // 44 padded to 48

// ---------- helpers ----------
__device__ __forceinline__ void split_bf16(float v, __bf16* hi, __bf16* lo) {
    __bf16 h = (__bf16)v;
    *hi = h;
    *lo = (__bf16)(v - (float)h);
}

__device__ __forceinline__ v8f wmma_bf16(v16bf a, v16bf b, v8f c) {
    // D = A(16x32) * B(32x16) + C, fp32 accumulate
    return __builtin_amdgcn_wmma_f32_16x16x32_bf16(
        false, a, false, b, (short)0, c, false, false);
}

__device__ __forceinline__ v16bf load_frag_a(const __bf16* p) {
    // A 16x32 bf16 layout: this lane's K-halves at p[0..7] and p[16..23]
    v8bf lo8 = *(const v8bf*)(p);
    v8bf hi8 = *(const v8bf*)(p + 16);
    v16bf f;
    #pragma unroll
    for (int i = 0; i < 8; ++i) { f[i] = lo8[i]; f[8 + i] = hi8[i]; }
    return f;
}

__device__ __forceinline__ v16bf load_frag_b(const __bf16* p) {
    // B 32x16 bf16 layout: this lane's K-halves at p[0..7] and p[16..23]
    v8bf lo8 = *(const v8bf*)(p);
    v8bf hi8 = *(const v8bf*)(p + 16);
    v16bf f;
    #pragma unroll
    for (int i = 0; i < 8; ++i) { f[i] = lo8[i]; f[8 + i] = hi8[i]; }
    return f;
}

// ======================================================================
// 3xBF16 WMMA GEMM:  C[M,N] = A[M,K] * W[N,K]^T (+ bias[N])
// A,W given as hi/lo bf16 planes (row-major).
// Each wave computes a 64x16 tile (4 M-tiles, B-fragments reused 4x,
// 4 independent accumulators hide the bf16 WMMA RAW hazard).
// store_bchw != 0: scatter rows m=(b*LL+l) into out[b][n][l] (final layout).
// ======================================================================
#define MT 4   // M-tiles per wave
__global__ __launch_bounds__(256)
void wmma_gemm_kernel(const __bf16* __restrict__ Ah, const __bf16* __restrict__ Al,
                      const __bf16* __restrict__ Wh, const __bf16* __restrict__ Wl,
                      const float* __restrict__ bias, float* __restrict__ C,
                      int M, int N, int K, int store_bchw)
{
    const int lane = threadIdx.x & 31;
    const int wave = threadIdx.x >> 5;
    const int n0 = blockIdx.x * 16;
    const int m0 = (blockIdx.y * 8 + wave) * (16 * MT);
    if (m0 >= M || n0 >= N) return;   // wave-uniform

    const int row_a = m0 + (lane & 15);
    const int row_b = n0 + (lane & 15);
    const int ka = (lane < 16) ? 0 : 8;    // A: K in {ka..ka+7, ka+16..ka+23}
    const int kb = (lane < 16) ? 0 : 8;    // B: K in {kb..kb+7, kb+16..kb+23}

    v8f acc[MT] = {};
    for (int k0 = 0; k0 < K; k0 += 32) {
        const __bf16* pbh = Wh + (size_t)row_b * K + k0 + kb;
        const __bf16* pbl = Wl + (size_t)row_b * K + k0 + kb;
        v16bf b_h = load_frag_b(pbh);
        v16bf b_l = load_frag_b(pbl);
        #pragma unroll
        for (int t = 0; t < MT; ++t) {
            const size_t ro = (size_t)(row_a + 16 * t) * K + k0 + ka;
            v16bf a_h = load_frag_a(Ah + ro);
            v16bf a_l = load_frag_a(Al + ro);
            acc[t] = wmma_bf16(a_h, b_h, acc[t]);   // hi*hi
            acc[t] = wmma_bf16(a_h, b_l, acc[t]);   // hi*lo
            acc[t] = wmma_bf16(a_l, b_h, acc[t]);   // lo*hi
        }
    }

    const int n  = n0 + (lane & 15);
    const float bv = bias ? bias[n] : 0.0f;
    #pragma unroll
    for (int t = 0; t < MT; ++t) {
        const int mb = m0 + 16 * t + ((lane >> 4) << 3);
        if (store_bchw) {
            #pragma unroll
            for (int r = 0; r < 8; ++r) {
                int m = mb + r;                       // m = b*LL + l
                int b = m >> 10, l = m & (LL - 1);
                C[((size_t)b * N + n) * LL + l] = acc[t][r] + bv;
            }
        } else {
            #pragma unroll
            for (int r = 0; r < 8; ++r)
                C[(size_t)(mb + r) * N + n] = acc[t][r] + bv;
        }
    }
}

// ======================================================================
// Prep kernels
// ======================================================================

// x [B,C,L] -> xT hi/lo bf16 planes [BL, C]
__global__ void prep_xT_kernel(const float* __restrict__ x,
                               __bf16* __restrict__ xh, __bf16* __restrict__ xl)
{
    int idx = blockIdx.x * blockDim.x + threadIdx.x;
    if (idx >= BL * CCH) return;
    int c = idx % CCH, bl = idx / CCH;
    int l = bl % LL, b = bl / LL;
    float v = x[((size_t)b * CCH + c) * LL + l];
    split_bf16(v, &xh[idx], &xl[idx]);
}

// Wc[j,c] = sum_o in_proj_w[j,o]*proj_w[o,c] ; bc[j] = sum_o in_proj_w[j,o]*proj_b[o]
__global__ void prep_comb_kernel(const float* __restrict__ in_proj_w,
                                 const float* __restrict__ proj_w,
                                 const float* __restrict__ proj_b,
                                 __bf16* __restrict__ Wch, __bf16* __restrict__ Wcl,
                                 float* __restrict__ bc)
{
    int idx = blockIdx.x * blockDim.x + threadIdx.x;
    if (idx >= NXZ * CCH) return;
    int c = idx % CCH, j = idx / CCH;
    float s = 0.f;
    for (int o = 0; o < CCH; ++o)
        s += in_proj_w[(size_t)j * CCH + o] * proj_w[(size_t)o * CCH + c];
    split_bf16(s, &Wch[idx], &Wcl[idx]);
    if (c == 0) {
        float sb = 0.f;
        for (int o = 0; o < CCH; ++o)
            sb += in_proj_w[(size_t)j * CCH + o] * proj_b[o];
        bc[j] = sb;
    }
}

// xproj_w [44,384] -> padded [48,384] hi/lo
__global__ void prep_xproj_kernel(const float* __restrict__ w,
                                  __bf16* __restrict__ wh, __bf16* __restrict__ wl)
{
    int idx = blockIdx.x * blockDim.x + threadIdx.x;
    if (idx >= NDBL * DIN) return;
    int r = idx / DIN, k = idx % DIN;
    float v = (r < (DTR + 2 * NST)) ? w[(size_t)r * DIN + k] : 0.f;
    split_bf16(v, &wh[idx], &wl[idx]);
}

// out_w [192,384] split
__global__ void prep_outw_kernel(const float* __restrict__ w,
                                 __bf16* __restrict__ wh, __bf16* __restrict__ wl)
{
    int idx = blockIdx.x * blockDim.x + threadIdx.x;
    if (idx >= CCH * DIN) return;
    split_bf16(w[idx], &wh[idx], &wl[idx]);
}

// A = -exp(A_log)
__global__ void prep_A_kernel(const float* __restrict__ A_log, float* __restrict__ A)
{
    int idx = blockIdx.x * blockDim.x + threadIdx.x;
    if (idx >= DIN * NST) return;
    A[idx] = -__expf(A_log[idx]);
}

// ======================================================================
// Causal depthwise conv (k=4) + SiLU, split result for GEMM2
// ======================================================================
__global__ void conv_silu_kernel(const float* __restrict__ xz,
                                 const float* __restrict__ conv_w,
                                 const float* __restrict__ conv_b,
                                 float* __restrict__ xc,
                                 __bf16* __restrict__ xch, __bf16* __restrict__ xcl)
{
    int idx = blockIdx.x * blockDim.x + threadIdx.x;
    if (idx >= BL * DIN) return;
    int d = idx % DIN, bl = idx / DIN;
    int l = bl % LL, b = bl / LL;
    float acc = conv_b[d];
    #pragma unroll
    for (int k = 0; k < 4; ++k) {
        int lp = l - 3 + k;
        if (lp >= 0)
            acc += conv_w[d * 4 + k] * xz[((size_t)b * LL + lp) * NXZ + d];
    }
    float s = acc / (1.f + __expf(-acc));   // SiLU
    xc[idx] = s;
    split_bf16(s, &xch[idx], &xcl[idx]);
}

// ======================================================================
// dt = softplus(dtr @ dtproj_w^T + dtproj_b), K = 12
// ======================================================================
__global__ void dt_kernel(const float* __restrict__ xdbl,
                          const float* __restrict__ dtproj_w,
                          const float* __restrict__ dtproj_b,
                          float* __restrict__ dt)
{
    int idx = blockIdx.x * blockDim.x + threadIdx.x;
    if (idx >= BL * DIN) return;
    int d = idx % DIN, bl = idx / DIN;
    float a = dtproj_b[d];
    #pragma unroll
    for (int r = 0; r < DTR; ++r)
        a += xdbl[(size_t)bl * NDBL + r] * dtproj_w[d * DTR + r];
    dt[idx] = (a > 20.f) ? a : log1pf(__expf(a));
}

// ======================================================================
// Selective scan (serial over L) + D-skip + SiLU(z) gate.
// One lane per (b,d): 16 states in registers, 1024 steps.
// ======================================================================
__global__ __launch_bounds__(128)
void scan_gate_kernel(const float* __restrict__ dt,
                      const float* __restrict__ A,      // [DIN,16], negative
                      const float* __restrict__ xdbl,   // [BL,48]: Bt=12..27, Ct=28..43
                      const float* __restrict__ xc,
                      const float* __restrict__ xz,     // z at cols 384..767
                      const float* __restrict__ Dp,
                      __bf16* __restrict__ yh, __bf16* __restrict__ yl)
{
    int g = blockIdx.x * blockDim.x + threadIdx.x;
    if (g >= BATCH * DIN) return;
    int d = g % DIN, b = g / DIN;

    float Ad[NST], h[NST];
    #pragma unroll
    for (int n = 0; n < NST; ++n) { Ad[n] = A[d * NST + n]; h[n] = 0.f; }
    const float Dv = Dp[d];

    for (int l = 0; l < LL; ++l) {
        const size_t bl = (size_t)b * LL + l;
        const float dtv = dt[bl * DIN + d];
        const float xv  = xc[bl * DIN + d];
        const float zv  = xz[bl * NXZ + DIN + d];
        const float* Bt = xdbl + bl * NDBL + DTR;
        const float* Ct = Bt + NST;
        const float dx = dtv * xv;
        float y = 0.f;
        #pragma unroll
        for (int n = 0; n < NST; ++n) {
            float dA = __expf(dtv * Ad[n]);
            h[n] = dA * h[n] + dx * Bt[n];
            y += h[n] * Ct[n];
        }
        y += xv * Dv;
        float gate = zv / (1.f + __expf(-zv));
        split_bf16(y * gate, &yh[bl * DIN + d], &yl[bl * DIN + d]);
    }
}

// ======================================================================
// Host side
// ======================================================================
extern "C" void kernel_launch(void* const* d_in, const int* in_sizes, int n_in,
                              void* d_out, int out_size, void* d_ws, size_t ws_size,
                              hipStream_t stream)
{
    const float* x         = (const float*)d_in[0];
    const float* proj_w    = (const float*)d_in[1];
    const float* proj_b    = (const float*)d_in[2];
    const float* in_proj_w = (const float*)d_in[3];
    const float* conv_w    = (const float*)d_in[4];
    const float* conv_b    = (const float*)d_in[5];
    const float* xproj_w   = (const float*)d_in[6];
    const float* dtproj_w  = (const float*)d_in[7];
    const float* dtproj_b  = (const float*)d_in[8];
    const float* A_log     = (const float*)d_in[9];
    const float* Dp        = (const float*)d_in[10];
    const float* out_w     = (const float*)d_in[11];
    float* out = (float*)d_out;

    // workspace carve-up (256B aligned)
    char* base = (char*)d_ws;
    size_t off = 0;
    auto alloc = [&](size_t bytes) -> void* {
        void* p = base + off;
        off = (off + bytes + 255) & ~(size_t)255;
        return p;
    };
    __bf16* xT_h    = (__bf16*)alloc((size_t)BL * CCH * 2);
    __bf16* xT_l    = (__bf16*)alloc((size_t)BL * CCH * 2);
    __bf16* Wc_h    = (__bf16*)alloc((size_t)NXZ * CCH * 2);
    __bf16* Wc_l    = (__bf16*)alloc((size_t)NXZ * CCH * 2);
    float*  bc      = (float*) alloc((size_t)NXZ * 4);
    float*  xz      = (float*) alloc((size_t)BL * NXZ * 4);
    float*  xc      = (float*) alloc((size_t)BL * DIN * 4);
    __bf16* xc_h    = (__bf16*)alloc((size_t)BL * DIN * 2);
    __bf16* xc_l    = (__bf16*)alloc((size_t)BL * DIN * 2);
    __bf16* xp_h    = (__bf16*)alloc((size_t)NDBL * DIN * 2);
    __bf16* xp_l    = (__bf16*)alloc((size_t)NDBL * DIN * 2);
    float*  xdbl    = (float*) alloc((size_t)BL * NDBL * 4);
    float*  dt      = (float*) alloc((size_t)BL * DIN * 4);
    float*  Aneg    = (float*) alloc((size_t)DIN * NST * 4);
    __bf16* ow_h    = (__bf16*)alloc((size_t)CCH * DIN * 2);
    __bf16* ow_l    = (__bf16*)alloc((size_t)CCH * DIN * 2);
    __bf16* y_h     = (__bf16*)alloc((size_t)BL * DIN * 2);
    __bf16* y_l     = (__bf16*)alloc((size_t)BL * DIN * 2);
    (void)ws_size; (void)in_sizes; (void)n_in; (void)out_size;

    auto blocks = [](int n, int t) { return (n + t - 1) / t; };
    const int MROWS = 16 * MT * 8;   // rows per GEMM block (8 waves x 64 rows)

    // 1) x transpose + split
    prep_xT_kernel<<<blocks(BL * CCH, 256), 256, 0, stream>>>(x, xT_h, xT_l);
    // 2) fold proj_in into in_proj
    prep_comb_kernel<<<blocks(NXZ * CCH, 256), 256, 0, stream>>>(
        in_proj_w, proj_w, proj_b, Wc_h, Wc_l, bc);
    // 3-5) small weight preps
    prep_xproj_kernel<<<blocks(NDBL * DIN, 256), 256, 0, stream>>>(xproj_w, xp_h, xp_l);
    prep_outw_kernel<<<blocks(CCH * DIN, 256), 256, 0, stream>>>(out_w, ow_h, ow_l);
    prep_A_kernel<<<blocks(DIN * NST, 256), 256, 0, stream>>>(A_log, Aneg);

    // 6) GEMM1: xz[BL,768] = xT[BL,192] * Wc^T + bc
    wmma_gemm_kernel<<<dim3(NXZ / 16, BL / MROWS), 256, 0, stream>>>(
        xT_h, xT_l, Wc_h, Wc_l, bc, xz, BL, NXZ, CCH, 0);

    // 7) causal conv + SiLU
    conv_silu_kernel<<<blocks(BL * DIN, 256), 256, 0, stream>>>(
        xz, conv_w, conv_b, xc, xc_h, xc_l);

    // 8) GEMM2: x_dbl[BL,48] = xc[BL,384] * xproj^T
    wmma_gemm_kernel<<<dim3(NDBL / 16, BL / MROWS), 256, 0, stream>>>(
        xc_h, xc_l, xp_h, xp_l, nullptr, xdbl, BL, NDBL, DIN, 0);

    // 9) dt projection + softplus
    dt_kernel<<<blocks(BL * DIN, 256), 256, 0, stream>>>(xdbl, dtproj_w, dtproj_b, dt);

    // 10) selective scan + gate
    scan_gate_kernel<<<blocks(BATCH * DIN, 128), 128, 0, stream>>>(
        dt, Aneg, xdbl, xc, xz, Dp, y_h, y_l);

    // 11) GEMM3: out[B,C,H,W] = y[BL,384] * out_w^T  (store fused with transpose)
    wmma_gemm_kernel<<<dim3(CCH / 16, BL / MROWS), 256, 0, stream>>>(
        y_h, y_l, ow_h, ow_l, nullptr, out, BL, CCH, DIN, 1);
}
